// MultiHeadCrossAttention_5497558139624
// MI455X (gfx1250) — matrix operand order repaired and checked by reading, not compiled, not run on hardware
//
#include <hip/hip_runtime.h>
#include <hip/hip_bf16.h>

// ---- CDNA5 WMMA types ----
typedef float  v2f    __attribute__((ext_vector_type(2)));
typedef float  v8f    __attribute__((ext_vector_type(8)));
typedef __bf16 v16bf  __attribute__((ext_vector_type(16)));
typedef __bf16 bf16x8 __attribute__((ext_vector_type(8)));
typedef __bf16 bf16x4 __attribute__((ext_vector_type(4)));
typedef int    v4i    __attribute__((ext_vector_type(4)));

__device__ __forceinline__ v8f wmma_f32x4(v2f a, v2f b, v8f c) {
    return __builtin_amdgcn_wmma_f32_16x16x4_f32(false, a, false, b, (short)0, c,
                                                 false, false);
}
__device__ __forceinline__ v8f wmma_bf16(v16bf a, v16bf b, v8f c) {
    return __builtin_amdgcn_wmma_f32_16x16x32_bf16(false, a, false, b, (short)0, c,
                                                   false, false);
}

// Async global->LDS copy (ASYNCcnt-tracked) when the toolchain declares it;
// plain b128 LDS copy otherwise. Builtin signature (from compiler diagnostic):
// (int4 addrspace(1)*, int4 addrspace(3)*, imm offset, imm cpol)
#if __has_builtin(__builtin_amdgcn_global_load_async_to_lds_b128) && \
    __has_builtin(__builtin_amdgcn_s_wait_asynccnt)
#define USE_ASYNC_LDS 1
#else
#define USE_ASYNC_LDS 0
#endif

__device__ __forceinline__ void copy16_g2l(const void* gsrc, void* ldst) {
#if USE_ASYNC_LDS
    __builtin_amdgcn_global_load_async_to_lds_b128(
        (__attribute__((address_space(1))) v4i*)gsrc,
        (__attribute__((address_space(3))) v4i*)ldst, 0, 0);
#else
    *(bf16x8*)ldst = *(const bf16x8*)gsrc;
#endif
}
__device__ __forceinline__ void async_join() {
#if USE_ASYNC_LDS
    __builtin_amdgcn_s_wait_asynccnt(0);
#endif
}

constexpr int Bn = 4;      // batch
constexpr int Tn = 1024;   // sequence
constexpr int En = 1024;   // embed
constexpr int Hn = 16;     // heads
constexpr int Dn = 64;     // head dim
constexpr float SM_SCALE = 0.125f;   // 1/sqrt(E/H) = 1/8

// fp32 -> RNE bf16 hi + bf16 residual lo (bf16x3 emulation split)
__device__ __forceinline__ void split4(const float4 v, bf16x4& hi, bf16x4& lo) {
    const float f[4] = {v.x, v.y, v.z, v.w};
    #pragma unroll
    for (int i = 0; i < 4; ++i) {
        const __bf16 h = (__bf16)f[i];
        hi[i] = h;
        lo[i] = (__bf16)(f[i] - (float)h);
    }
}

__device__ __forceinline__ v16bf frag16(const __bf16* p0, const __bf16* p1) {
    const bf16x8 c0 = *(const bf16x8*)p0;
    const bf16x8 c1 = *(const bf16x8*)p1;
    return __builtin_shufflevector(c0, c1, 0, 1, 2, 3, 4, 5, 6, 7,
                                           8, 9, 10, 11, 12, 13, 14, 15);
}

// ============================================================================
// Kernel 0: one-time fp32 -> bf16 hi/lo split of an array (n % 1024 == 0).
// ============================================================================
__global__ __launch_bounds__(256)
void split_kernel(const float* __restrict__ src, __bf16* __restrict__ hi,
                  __bf16* __restrict__ lo)
{
    const size_t i4 = (size_t)blockIdx.x * 256 + threadIdx.x;
    const float4 v = *(const float4*)(src + i4 * 4);
    bf16x4 h, l;
    split4(v, h, l);
    *(bf16x4*)(hi + i4 * 4) = h;
    *(bf16x4*)(lo + i4 * 4) = l;
}

// ============================================================================
// Shared WMMA compute stage for the bf16x3 GEMM cores: per wave, one 16-row
// A fragment pair vs 4 N-subtiles, 3 passes (hh + hl + lh), fp32 accumulate.
// ============================================================================
__device__ __forceinline__ void bf16x3_mma(
    const __bf16 (&Ash)[64][40], const __bf16 (&Asl)[64][40],
    const __bf16 (&Bsh)[128][40], const __bf16 (&Bsl)[128][40],
    int m, int khalf, int koff2, int wx, int lane, v8f (&acc)[4])
{
    const v16bf ah = frag16(&Ash[m][koff2], &Ash[m][koff2 + 8]);
    const v16bf al = frag16(&Asl[m][koff2], &Asl[m][koff2 + 8]);
    v16bf bh[4], bl[4];
    #pragma unroll
    for (int nt = 0; nt < 4; ++nt) {
        const int n = wx * 64 + nt * 16 + (lane & 15);
        bh[nt] = frag16(&Bsh[n][koff2], &Bsh[n][koff2 + 8]);
        bl[nt] = frag16(&Bsl[n][koff2], &Bsl[n][koff2 + 8]);
    }
    #pragma unroll
    for (int nt = 0; nt < 4; ++nt) acc[nt] = wmma_bf16(ah, bh[nt], acc[nt]);
    #pragma unroll
    for (int nt = 0; nt < 4; ++nt) acc[nt] = wmma_bf16(ah, bl[nt], acc[nt]);
    #pragma unroll
    for (int nt = 0; nt < 4; ++nt) acc[nt] = wmma_bf16(al, bh[nt], acc[nt]);
}

// ============================================================================
// GEMM core (qkv): A fp32 (split in-kernel, 2 float4/thread), W pre-split
// bf16 hi/lo (pure 16B copies, async-LDS capable). C(64x128), BK=32.
// ============================================================================
__device__ __forceinline__ void gemm_Af32_Bpre(
    const float* __restrict__ A,
    const __bf16* __restrict__ Wh, const __bf16* __restrict__ Wl,
    int m0, int n0,
    __bf16 (&Ash)[64][40],  __bf16 (&Asl)[64][40],
    __bf16 (&Bsh)[128][40], __bf16 (&Bsl)[128][40], v8f (&acc)[4])
{
    const int tid   = threadIdx.x;
    const int lane  = tid & 31;
    const int wave  = tid >> 5;
    const int wy    = wave & 3;
    const int wx    = wave >> 2;
    const int m     = wy * 16 + (lane & 15);
    const int khalf = (lane >= 16) ? 8 : 0;
    const int koff2 = (lane >= 16) ? 16 : 0;

    v8f z = {};
    #pragma unroll
    for (int j = 0; j < 4; ++j) acc[j] = z;

    for (int k0 = 0; k0 < En; k0 += 32) {
        __syncthreads();
        #pragma unroll
        for (int i = 0; i < 2; ++i) {   // A: stage + split 64x32 fp32
            const int idx = tid + i * 256;
            const int r = idx >> 3, cg = idx & 7;
            const float4 v = *(const float4*)(A + (size_t)(m0 + r) * En + k0 + cg * 4);
            bf16x4 h, l;
            split4(v, h, l);
            *(bf16x4*)&Ash[r][cg * 4] = h;
            *(bf16x4*)&Asl[r][cg * 4] = l;
        }
        #pragma unroll
        for (int i = 0; i < 2; ++i) {   // W: pure copy of pre-split 128x32 bf16
            const int idx = tid + i * 256;
            const int r = idx >> 2, ch = idx & 3;
            const size_t g = (size_t)(n0 + r) * En + k0 + ch * 8;
            copy16_g2l(Wh + g, &Bsh[r][ch * 8]);
            copy16_g2l(Wl + g, &Bsl[r][ch * 8]);
        }
        if (k0 + 32 < En) {
            __builtin_prefetch(A  + (size_t)(m0 + (tid >> 2)) * En + k0 + 32, 0, 0);
            __builtin_prefetch(Wh + (size_t)(n0 + (tid >> 1)) * En + k0 + 32, 0, 0);
        }
        async_join();
        __syncthreads();
        bf16x3_mma(Ash, Asl, Bsh, Bsl, m, khalf, koff2, wx, lane, acc);
    }
}

// ============================================================================
// GEMM core (out-proj): both A and W pre-split bf16 hi/lo — staging is 100%
// 16B copies (async-LDS capable), zero conversions in the hot loop.
// ============================================================================
__device__ __forceinline__ void gemm_pre(
    const __bf16* __restrict__ Ah, const __bf16* __restrict__ Al,
    const __bf16* __restrict__ Wh, const __bf16* __restrict__ Wl,
    int m0, int n0,
    __bf16 (&Ash)[64][40],  __bf16 (&Asl)[64][40],
    __bf16 (&Bsh)[128][40], __bf16 (&Bsl)[128][40], v8f (&acc)[4])
{
    const int tid   = threadIdx.x;
    const int lane  = tid & 31;
    const int wave  = tid >> 5;
    const int wy    = wave & 3;
    const int wx    = wave >> 2;
    const int m     = wy * 16 + (lane & 15);
    const int khalf = (lane >= 16) ? 8 : 0;
    const int koff2 = (lane >= 16) ? 16 : 0;

    v8f z = {};
    #pragma unroll
    for (int j = 0; j < 4; ++j) acc[j] = z;

    for (int k0 = 0; k0 < En; k0 += 32) {
        __syncthreads();
        {   // A: 64x32 bf16 hi+lo (256 chunks each, 1+1 per thread)
            const int r = tid >> 2, ch = tid & 3;
            const size_t g = (size_t)(m0 + r) * En + k0 + ch * 8;
            copy16_g2l(Ah + g, &Ash[r][ch * 8]);
            copy16_g2l(Al + g, &Asl[r][ch * 8]);
        }
        #pragma unroll
        for (int i = 0; i < 2; ++i) {   // W: 128x32 bf16 hi+lo
            const int idx = tid + i * 256;
            const int r = idx >> 2, ch = idx & 3;
            const size_t g = (size_t)(n0 + r) * En + k0 + ch * 8;
            copy16_g2l(Wh + g, &Bsh[r][ch * 8]);
            copy16_g2l(Wl + g, &Bsl[r][ch * 8]);
        }
        if (k0 + 32 < En) {
            __builtin_prefetch(Ah + (size_t)(m0 + (tid >> 2)) * En + k0 + 32, 0, 0);
            __builtin_prefetch(Wh + (size_t)(n0 + (tid >> 1)) * En + k0 + 32, 0, 0);
        }
        async_join();
        __syncthreads();
        bf16x3_mma(Ash, Asl, Bsh, Bsl, m, khalf, koff2, wx, lane, acc);
    }
}

// ============================================================================
// Kernel 1: fused QKV projection + scatter into [s][b][h][t][d] fp32 Q/K/V.
// qkv row r = d*48 + k*16 + h. grid = (24, 16, 8), block = 256
// ============================================================================
__global__ __launch_bounds__(256)
void qkv_proj_kernel(const float* __restrict__ x, const float* __restrict__ y,
                     const __bf16* __restrict__ Wh, const __bf16* __restrict__ Wl,
                     float* __restrict__ Qw, float* __restrict__ Kw,
                     float* __restrict__ Vw)
{
    __shared__ __align__(16) __bf16 Ash[64][40];
    __shared__ __align__(16) __bf16 Asl[64][40];
    __shared__ __align__(16) __bf16 Bsh[128][40];
    __shared__ __align__(16) __bf16 Bsl[128][40];

    const int s = blockIdx.z >> 2;
    const int b = blockIdx.z & 3;
    const float* A = (s ? y : x) + (size_t)b * Tn * En;
    const __bf16* wh = Wh + (size_t)s * 3 * En * En;
    const __bf16* wl = Wl + (size_t)s * 3 * En * En;
    const int m0 = blockIdx.y * 64;    // t
    const int n0 = blockIdx.x * 128;   // r in [0,3072)

    v8f acc[4];
    gemm_Af32_Bpre(A, wh, wl, m0, n0, Ash, Asl, Bsh, Bsl, acc);

    const int lane = threadIdx.x & 31;
    const int wave = threadIdx.x >> 5;
    const int wy = wave & 3, wx = wave >> 2;
    const int hi8 = (lane >= 16) ? 8 : 0;

    #pragma unroll
    for (int nt = 0; nt < 4; ++nt) {
        const int r   = n0 + wx * 64 + nt * 16 + (lane & 15);
        const int d   = r / 48;
        const int rem = r - d * 48;
        const int kq  = rem >> 4;      // 0=Q, 1=K, 2=V
        const int h   = rem & 15;
        float* dst = (kq == 0) ? Qw : (kq == 1 ? Kw : Vw);
        const size_t base = ((((size_t)s * Bn + b) * Hn + h) * Tn) * Dn + d;
        #pragma unroll
        for (int j = 0; j < 8; ++j) {
            const int t = m0 + wy * 16 + j + hi8;
            dst[base + (size_t)t * Dn] = acc[nt][j];
        }
    }
}

// ============================================================================
// Kernel 2: flash cross-attention, exact fp32 WMMA; epilogue emits O as
// pre-split bf16 hi/lo for the out-projection. grid = (16,16,8), block = 128
// ============================================================================
__global__ __launch_bounds__(128)
void cross_attn_kernel(const float* __restrict__ Qw, const float* __restrict__ Kw,
                       const float* __restrict__ Vw,
                       __bf16* __restrict__ Ohi, __bf16* __restrict__ Olo)
{
    __shared__ __align__(16) float Qs[64][68];
    __shared__ __align__(16) float Ks[16][68];
    __shared__ __align__(16) float Vs[16][68];
    __shared__ float Ss[4][16][17];
    __shared__ float rowm[4][16], rowl[4][16], rowsc[4][16];

    const int q0 = blockIdx.x * 64;
    const int h  = blockIdx.y;
    const int s  = blockIdx.z >> 2;              // output stream
    const int b  = blockIdx.z & 3;
    const int qs = 1 - s;                        // query stream

    const size_t headQ  = (((size_t)qs * Bn + b) * Hn + h) * (size_t)Tn * Dn;
    const size_t headKV = (((size_t)s  * Bn + b) * Hn + h) * (size_t)Tn * Dn;
    const float* Qg = Qw + headQ;
    const float* Kg = Kw + headKV;
    const float* Vg = Vw + headKV;

    const int tid   = threadIdx.x;
    const int lane  = tid & 31;
    const int wave  = tid >> 5;
    const int hi8   = (lane >= 16) ? 8 : 0;
    const int koffL = (lane >= 16) ? 2 : 0;

    #pragma unroll
    for (int i = 0; i < 8; ++i) {
        const int idx = tid + i * 128;
        const int r = idx >> 4, cg = idx & 15;
        *(float4*)&Qs[r][cg * 4] =
            *(const float4*)(Qg + (size_t)(q0 + r) * Dn + cg * 4);
    }
    if (lane < 16) {
        rowm[wave][lane] = -3.0e38f;
        rowl[wave][lane] = 0.0f;
    }
    v8f z = {};
    v8f oacc[4];
    #pragma unroll
    for (int nt = 0; nt < 4; ++nt) oacc[nt] = z;
    __syncthreads();

    for (int jt = 0; jt < Tn / 16; ++jt) {
        __syncthreads();
        #pragma unroll
        for (int i = 0; i < 4; ++i) {
            const int idx = tid + i * 128;
            if (idx < 256) {
                const int r = idx >> 4, cg = idx & 15;
                *(float4*)&Ks[r][cg * 4] =
                    *(const float4*)(Kg + (size_t)(jt * 16 + r) * Dn + cg * 4);
            } else {
                const int i2 = idx - 256;
                const int r = i2 >> 4, cg = i2 & 15;
                *(float4*)&Vs[r][cg * 4] =
                    *(const float4*)(Vg + (size_t)(jt * 16 + r) * Dn + cg * 4);
            }
        }
        __syncthreads();

        v8f sacc = z;
        #pragma unroll
        for (int kk = 0; kk < 16; ++kk) {
            const int ko = kk * 4 + koffL;
            v2f a;  a.x  = Qs[wave * 16 + (lane & 15)][ko];
                    a.y  = Qs[wave * 16 + (lane & 15)][ko + 1];
            v2f bb; bb.x = Ks[lane & 15][ko];
                    bb.y = Ks[lane & 15][ko + 1];
            sacc = wmma_f32x4(a, bb, sacc);
        }
        #pragma unroll
        for (int j = 0; j < 8; ++j)
            Ss[wave][j + hi8][lane & 15] = sacc[j] * SM_SCALE;
        __syncthreads();

        if (lane < 16) {
            const int r = lane;
            const float mold = rowm[wave][r];
            float tm = mold;
            #pragma unroll
            for (int c = 0; c < 16; ++c) tm = fmaxf(tm, Ss[wave][r][c]);
            const float sc = __expf(mold - tm);
            float sum = 0.0f;
            #pragma unroll
            for (int c = 0; c < 16; ++c) {
                const float p = __expf(Ss[wave][r][c] - tm);
                Ss[wave][r][c] = p;
                sum += p;
            }
            rowl[wave][r]  = rowl[wave][r] * sc + sum;
            rowm[wave][r]  = tm;
            rowsc[wave][r] = sc;
        }
        __syncthreads();

        float f[8];
        #pragma unroll
        for (int j = 0; j < 8; ++j) f[j] = rowsc[wave][j + hi8];
        #pragma unroll
        for (int nt = 0; nt < 4; ++nt)
            #pragma unroll
            for (int j = 0; j < 8; ++j) oacc[nt][j] *= f[j];

        #pragma unroll
        for (int kk = 0; kk < 4; ++kk) {
            const int ko = kk * 4 + koffL;
            v2f a; a.x = Ss[wave][lane & 15][ko];
                   a.y = Ss[wave][lane & 15][ko + 1];
            #pragma unroll
            for (int nt = 0; nt < 4; ++nt) {
                const int nc = nt * 16 + (lane & 15);
                v2f bb; bb.x = Vs[ko][nc]; bb.y = Vs[ko + 1][nc];
                oacc[nt] = wmma_f32x4(a, bb, oacc[nt]);
            }
        }
    }

    // Normalize; store O pre-split as bf16 hi/lo into [s][b][t][h*64+d]
    float inv[8];
    #pragma unroll
    for (int j = 0; j < 8; ++j) inv[j] = 1.0f / rowl[wave][j + hi8];
    #pragma unroll
    for (int nt = 0; nt < 4; ++nt) {
        const int e = h * 64 + nt * 16 + (lane & 15);
        #pragma unroll
        for (int j = 0; j < 8; ++j) {
            const int t = q0 + wave * 16 + j + hi8;
            const size_t oi = (((size_t)s * Bn + b) * Tn + t) * En + e;
            const float val = oacc[nt][j] * inv[j];
            const __bf16 hb = (__bf16)val;
            Ohi[oi] = hb;
            Olo[oi] = (__bf16)(val - (float)hb);
        }
    }
}

// ============================================================================
// Kernel 3: out_s = O_s (4096x1024) @ Wout_s^T, fully pre-split bf16x3.
// grid = (8, 64, 2), block = 256
// ============================================================================
__global__ __launch_bounds__(256)
void out_proj_kernel(const __bf16* __restrict__ Ohi, const __bf16* __restrict__ Olo,
                     const __bf16* __restrict__ Wh,  const __bf16* __restrict__ Wl,
                     float* __restrict__ out)
{
    __shared__ __align__(16) __bf16 Ash[64][40];
    __shared__ __align__(16) __bf16 Asl[64][40];
    __shared__ __align__(16) __bf16 Bsh[128][40];
    __shared__ __align__(16) __bf16 Bsl[128][40];

    const int s = blockIdx.z;
    const __bf16* Ah = Ohi + (size_t)s * Bn * Tn * En;
    const __bf16* Al = Olo + (size_t)s * Bn * Tn * En;
    const __bf16* wh = Wh  + (size_t)s * En * En;
    const __bf16* wl = Wl  + (size_t)s * En * En;
    float* C = out + (size_t)s * Bn * Tn * En;
    const int m0 = blockIdx.y * 64;
    const int n0 = blockIdx.x * 128;

    v8f acc[4];
    gemm_pre(Ah, Al, wh, wl, m0, n0, Ash, Asl, Bsh, Bsl, acc);

    const int lane = threadIdx.x & 31;
    const int wave = threadIdx.x >> 5;
    const int wy = wave & 3, wx = wave >> 2;
    const int hi8 = (lane >= 16) ? 8 : 0;

    #pragma unroll
    for (int nt = 0; nt < 4; ++nt) {
        const int col = n0 + wx * 64 + nt * 16 + (lane & 15);
        #pragma unroll
        for (int j = 0; j < 8; ++j) {
            const int row = m0 + wy * 16 + j + hi8;
            C[(size_t)row * En + col] = acc[nt][j];
        }
    }
}

// ============================================================================
extern "C" void kernel_launch(void* const* d_in, const int* in_sizes, int n_in,
                              void* d_out, int out_size, void* d_ws, size_t ws_size,
                              hipStream_t stream) {
    (void)in_sizes; (void)n_in; (void)out_size; (void)ws_size;

    const float* x     = (const float*)d_in[0];
    const float* y     = (const float*)d_in[1];
    const float* Wqkv1 = (const float*)d_in[2];
    const float* Wqkv2 = (const float*)d_in[3];
    const float* Wout1 = (const float*)d_in[4];
    const float* Wout2 = (const float*)d_in[5];
    float* out = (float*)d_out;

    // Workspace map (160 MB):
    //   Q,K,V fp32: 3 x 8M floats (96 MB)
    //   O hi/lo bf16: 2 x 8M (32 MB)
    //   Wqkv hi/lo bf16: 2 x 6M (24 MB)
    //   Wout hi/lo bf16: 2 x 2M (8 MB)
    const size_t SZQ = (size_t)2 * Bn * Hn * Tn * Dn;   // 8,388,608
    const size_t SZW = (size_t)3 * En * En;             // 3,145,728 per stream
    const size_t SZO = (size_t)En * En;                 // 1,048,576 per stream
    float*  qws = (float*)d_ws;
    float*  kws = qws + SZQ;
    float*  vws = kws + SZQ;
    __bf16* ohi = (__bf16*)(vws + SZQ);
    __bf16* olo = ohi + SZQ;
    __bf16* wqh = olo + SZQ;
    __bf16* wql = wqh + 2 * SZW;
    __bf16* woh = wql + 2 * SZW;
    __bf16* wol = woh + 2 * SZO;

    // One-time weight splits (reads 32 MB once; removes cvt work from GEMMs)
    split_kernel<<<dim3(SZW / 1024), 256, 0, stream>>>(Wqkv1, wqh, wql);
    split_kernel<<<dim3(SZW / 1024), 256, 0, stream>>>(Wqkv2, wqh + SZW, wql + SZW);
    split_kernel<<<dim3(SZO / 1024), 256, 0, stream>>>(Wout1, woh, wol);
    split_kernel<<<dim3(SZO / 1024), 256, 0, stream>>>(Wout2, woh + SZO, wol + SZO);

    qkv_proj_kernel<<<dim3(3 * En / 128, Tn / 64, 2 * Bn), 256, 0, stream>>>(
        x, y, wqh, wql, qws, kws, vws);
    cross_attn_kernel<<<dim3(Tn / 64, Hn, 2 * Bn), 128, 0, stream>>>(
        qws, kws, vws, ohi, olo);
    out_proj_kernel<<<dim3(En / 128, Bn * Tn / 64, 2), 256, 0, stream>>>(
        ohi, olo, woh, wol, out);
}
